// MultiHeadAttention_44315472560265
// MI455X (gfx1250) — matrix operand order repaired and checked by reading, not compiled, not run on hardware
//
#include <hip/hip_runtime.h>

#define D_MODEL 1024
#define NB 4
#define TSEQ 2048
#define NHEADS 16
#define HDIM 64
#define BT (NB * TSEQ)

typedef __attribute__((ext_vector_type(16))) __bf16 v16bf;
typedef __attribute__((ext_vector_type(8)))  float  v8f;

typedef unsigned int tdm_v4u __attribute__((ext_vector_type(4)));
typedef int          tdm_v8i __attribute__((ext_vector_type(8)));
typedef int          tdm_v4i __attribute__((ext_vector_type(4)));

#if defined(__has_builtin)
#if __has_builtin(__builtin_amdgcn_tensor_load_to_lds) && \
    __has_builtin(__builtin_amdgcn_s_wait_tensorcnt)
#define USE_TDM 1
#endif
#endif
#ifndef USE_TDM
#define USE_TDM 0
#endif

union AFrag {
  unsigned int u[8];
  v16bf v;
};

union U4 {
  uint4 q;
  unsigned short s[8];
};

struct BF4 { unsigned short a, b, c, d; };

__device__ __forceinline__ unsigned short f2bf(float f) {
  unsigned int u = __float_as_uint(f);
  u += 0x7FFFu + ((u >> 16) & 1u);   // round-to-nearest-even
  return (unsigned short)(u >> 16);
}

// A-matrix (16-bit, 16x32) K index for VGPR-pair p, half hi (ISA 7.12.2)
__device__ __forceinline__ int a_kk(int p, int hi) {
  return ((p < 4) ? 0 : 16) + hi * 8 + (p & 3) * 2;
}
// B-matrix (16-bit, 32x16) K index for VGPR-pair p, half hi (same order as A)
__device__ __forceinline__ int b_kk(int p, int hi) {
  return ((p < 4) ? 0 : 16) + hi * 8 + (p & 3) * 2;
}

// LDS byte offset of a __shared__ pointer: low 32 bits of the flat address
// (flat LDS addr = {shared-aperture base, lds_offset}, ISA 10.2).
__device__ __forceinline__ unsigned lds_off(const void* p) {
  return (unsigned)(unsigned long long)p;
}

// ---------------------------------------------------------------------------
// TDM: async 2D bf16 tile load (Global -> LDS), Tensor DMA descriptor per
// ISA ch. 8.  tile_w elements per row, tile_h rows, row stride in elements.
// Tensor dims set huge (1<<20) so no OOB clamping occurs.
// This toolchain's builtin takes 6 args (g0, g1, g2, g3, g_extra, cpol).
// ---------------------------------------------------------------------------
__device__ __forceinline__ void tdm_load_2d(const void* gptr, unsigned ldsoff,
                                            unsigned tile_w, unsigned tile_h,
                                            unsigned row_stride) {
#if USE_TDM
  unsigned long long ga = (unsigned long long)gptr;
  tdm_v4u g0;
  g0[0] = 1u;                                   // count=1, no gather/iterate
  g0[1] = ldsoff;                               // lds_addr [63:32]
  g0[2] = (unsigned)ga;                         // global_addr [95:64]
  g0[3] = (unsigned)(ga >> 32) | (2u << 30);    // global_addr hi | type=2
  tdm_v8i g1;
  g1[0] = (int)(1u << 16);                      // data_size=1 (2 bytes)
  g1[1] = 0;                                    // tensor_dim0 = 1<<20 (lo16=0)
  g1[2] = 16;                                   // dim0 hi | tensor_dim1 lo16=0
  g1[3] = (int)(16u | (tile_w << 16));          // dim1 hi | tile_dim0
  g1[4] = (int)tile_h;                          // tile_dim1, tile_dim2=0
  g1[5] = (int)row_stride;                      // tensor_dim0_stride lo32
  g1[6] = 0;
  g1[7] = 0;
  tdm_v4i z4 = {0, 0, 0, 0};
  tdm_v8i z8 = {0, 0, 0, 0, 0, 0, 0, 0};
  __builtin_amdgcn_tensor_load_to_lds(g0, g1, z4, z4, z8, 0);
#endif
}

__device__ __forceinline__ void tdm_wait() {
#if USE_TDM
  __builtin_amdgcn_s_wait_tensorcnt(0);
#endif
}

// Fallback synchronous tile copy (bf16 src), 16B chunks.
__device__ __forceinline__ void copy_tile_bf16(
    const unsigned short* __restrict__ g, unsigned short* __restrict__ s,
    int tile_w, int tile_h, int row_stride, int tid, int nthreads) {
  int chunks = (tile_w * tile_h) >> 3;
  int cpr = tile_w >> 3;                        // 16B chunks per row
  for (int i = tid; i < chunks; i += nthreads) {
    int r = i / cpr, c = (i % cpr) * 8;
    *(uint4*)&s[r * tile_w + c] =
        *(const uint4*)(g + (size_t)r * row_stride + c);
  }
}

// ---------------------------------------------------------------------------
// Kernel 0a: fp32 -> bf16 (flat, float4 granularity)
// ---------------------------------------------------------------------------
__global__ __launch_bounds__(256) void cvt_bf16(
    const float* __restrict__ src, unsigned short* __restrict__ dst, int n4) {
  int i = blockIdx.x * blockDim.x + threadIdx.x;
  if (i < n4) {
    float4 f = ((const float4*)src)[i];
    BF4 b = { f2bf(f.x), f2bf(f.y), f2bf(f.z), f2bf(f.w) };
    ((BF4*)dst)[i] = b;
  }
}

// ---------------------------------------------------------------------------
// Kernel 0b: fp32 W[k][n] -> bf16 Wt[n][k] (32x32 LDS tiles, both coalesced)
// ---------------------------------------------------------------------------
__global__ __launch_bounds__(256) void cvt_bf16_T(
    const float* __restrict__ W, unsigned short* __restrict__ Wt) {
  __shared__ float tile[32][33];
  const int n0 = blockIdx.x * 32, k0 = blockIdx.y * 32;
  const int tx = threadIdx.x & 31, ty = threadIdx.x >> 5;
#pragma unroll
  for (int r = ty; r < 32; r += 8)
    tile[r][tx] = W[(size_t)(k0 + r) * D_MODEL + n0 + tx];
  __syncthreads();
#pragma unroll
  for (int r = ty; r < 32; r += 8)
    Wt[(size_t)(n0 + r) * D_MODEL + k0 + tx] = f2bf(tile[tx][r]);
}

// ---------------------------------------------------------------------------
// Kernel 1: fused QKV projection (bf16 in, bf16 out).
// Block tile 128x64, 256 threads (8 waves), K-step 32, TDM-staged tiles.
// ---------------------------------------------------------------------------
__global__ __launch_bounds__(256) void qkv_gemm(
    const unsigned short* __restrict__ xb,
    const unsigned short* __restrict__ Wqt,
    const unsigned short* __restrict__ Wkt,
    const unsigned short* __restrict__ Wvt,
    unsigned short* __restrict__ qb,
    unsigned short* __restrict__ kb,
    unsigned short* __restrict__ vb)
{
  __shared__ __align__(16) unsigned short sA[128 * 32];   // x tile [row][k]
  __shared__ __align__(16) unsigned short sBT[64 * 32];   // Wt tile [n][k]

  const int tid  = threadIdx.x;
  const int wave = tid >> 5;
  const int lane = tid & 31;
  const int m    = lane & 15;
  const int hi   = lane >> 4;

  const int row0 = blockIdx.x * 128;
  const int ncol = blockIdx.y * 64;
  const int which = ncol >> 10;
  const int wcol  = ncol & 1023;
  const unsigned short* Wt = (which == 0) ? Wqt : (which == 1) ? Wkt : Wvt;
  unsigned short* outp = (which == 0) ? qb : (which == 1) ? kb : vb;

  const unsigned ldsA = lds_off(&sA[0]);
  const unsigned ldsB = lds_off(&sBT[0]);

  v8f acc[4];
#pragma unroll
  for (int j = 0; j < 4; j++) acc[j] = (v8f){0.f,0.f,0.f,0.f,0.f,0.f,0.f,0.f};

  for (int k0 = 0; k0 < D_MODEL; k0 += 32) {
    __syncthreads();
#if USE_TDM
    if (tid < 32) {                        // wave 0 drives the Tensor DMA
      tdm_load_2d(xb + (size_t)row0 * D_MODEL + k0, ldsA, 32, 128, D_MODEL);
      tdm_load_2d(Wt + (size_t)wcol * D_MODEL + k0, ldsB, 32, 64, D_MODEL);
      if (k0 + 32 < D_MODEL)
        __builtin_prefetch(Wt + (size_t)wcol * D_MODEL + k0 + 32, 0, 1);
      tdm_wait();
    }
#else
    copy_tile_bf16(xb + (size_t)row0 * D_MODEL + k0, sA, 32, 128, D_MODEL, tid, 256);
    copy_tile_bf16(Wt + (size_t)wcol * D_MODEL + k0, sBT, 32, 64, D_MODEL, tid, 256);
#endif
    __syncthreads();

    AFrag a;
    const unsigned short* ap = &sA[(wave * 16 + m) * 32];
#pragma unroll
    for (int p = 0; p < 8; p++)
      a.u[p] = *(const unsigned int*)(ap + a_kk(p, hi));
#pragma unroll
    for (int j = 0; j < 4; j++) {
      AFrag b;
      const unsigned short* bp = &sBT[(j * 16 + m) * 32];
#pragma unroll
      for (int p = 0; p < 8; p++)
        b.u[p] = *(const unsigned int*)(bp + b_kk(p, hi));
      acc[j] = __builtin_amdgcn_wmma_f32_16x16x32_bf16(
          false, a.v, false, b.v, (short)0, acc[j], false, false);
    }
  }

#pragma unroll
  for (int j = 0; j < 4; j++)
#pragma unroll
    for (int r = 0; r < 8; r++) {
      int grow = row0 + wave * 16 + r + hi * 8;
      int gcol = wcol + j * 16 + m;
      outp[(size_t)grow * D_MODEL + gcol] = f2bf(acc[j][r]);
    }
}

// ---------------------------------------------------------------------------
// Kernel 2: causal flash attention per (batch, head, 64-row q tile).
// 128 threads = 4 waves; KV tile 64; K/V tiles staged by TDM.
// ---------------------------------------------------------------------------
__global__ __launch_bounds__(128) void flash_attn(
    const unsigned short* __restrict__ qb,
    const unsigned short* __restrict__ kb,
    const unsigned short* __restrict__ vb,
    unsigned short* __restrict__ ctxb)
{
  __shared__ __align__(16) unsigned short sK[64 * 64];     // K tile [kv][d]
  __shared__ __align__(16) unsigned short sVraw[64 * 64];  // V tile [kv][d]
  __shared__ __align__(16) unsigned short sVt[64 * 64];    // V tile [d][kv]
  __shared__ __align__(16) unsigned short sP[4 * 16 * 64]; // per-wave P staging

  const int tid  = threadIdx.x;
  const int wave = tid >> 5;
  const int lane = tid & 31;
  const int m    = lane & 15;
  const int hi   = lane >> 4;

  const int bb = blockIdx.z;
  const int h  = blockIdx.y;
  const int qrow0 = blockIdx.x * 64 + wave * 16;
  unsigned short* sPw = &sP[wave * 16 * 64];

  const unsigned ldsK  = lds_off(&sK[0]);
  const unsigned ldsVr = lds_off(&sVraw[0]);

  // ---- preload Q fragments (16 x 64 -> two 16x32 A frags) ----
  AFrag aq[2];
  {
    const unsigned short* qp =
        qb + ((size_t)bb * TSEQ + qrow0 + m) * D_MODEL + h * HDIM;
#pragma unroll
    for (int s = 0; s < 2; s++)
#pragma unroll
      for (int p = 0; p < 8; p++)
        aq[s].u[p] = *(const unsigned int*)(qp + s * 32 + a_kk(p, hi));
  }

  v8f o[4];
#pragma unroll
  for (int j = 0; j < 4; j++) o[j] = (v8f){0.f,0.f,0.f,0.f,0.f,0.f,0.f,0.f};
  float mrow[8], lrow[8];
#pragma unroll
  for (int r = 0; r < 8; r++) { mrow[r] = -3.0e38f; lrow[r] = 0.f; }

  const int ktmax = blockIdx.x;
  for (int kt = 0; kt <= ktmax; kt++) {
    __syncthreads();
    const unsigned short* kg =
        kb + ((size_t)bb * TSEQ + kt * 64) * D_MODEL + h * HDIM;
    const unsigned short* vg =
        vb + ((size_t)bb * TSEQ + kt * 64) * D_MODEL + h * HDIM;
#if USE_TDM
    if (tid < 32) {
      tdm_load_2d(kg, ldsK, 64, 64, D_MODEL);
      tdm_load_2d(vg, ldsVr, 64, 64, D_MODEL);
      tdm_wait();
    }
#else
    copy_tile_bf16(kg, sK, 64, 64, D_MODEL, tid, 128);
    copy_tile_bf16(vg, sVraw, 64, 64, D_MODEL, tid, 128);
#endif
    __syncthreads();

    // ---- transpose V in LDS: [kv][d] -> [d][kv] ----
#pragma unroll
    for (int ii = 0; ii < 4; ii++) {
      int i  = tid + ii * 128;
      int kv = i >> 3;
      int d0 = (i & 7) * 8;
      U4 u;
      u.q = *(const uint4*)&sVraw[kv * 64 + d0];
#pragma unroll
      for (int jj = 0; jj < 8; jj++) sVt[(d0 + jj) * 64 + kv] = u.s[jj];
    }
    __syncthreads();

    // ---- scores: S = Q * K^T (16q x 64kv per wave) ----
    v8f s_acc[4];
#pragma unroll
    for (int j = 0; j < 4; j++) s_acc[j] = (v8f){0.f,0.f,0.f,0.f,0.f,0.f,0.f,0.f};
#pragma unroll
    for (int s = 0; s < 2; s++) {
#pragma unroll
      for (int j = 0; j < 4; j++) {
        AFrag bk;
        const unsigned short* kp = &sK[(j * 16 + m) * 64 + s * 32];
#pragma unroll
        for (int p = 0; p < 8; p++)
          bk.u[p] = *(const unsigned int*)(kp + b_kk(p, hi));
        s_acc[j] = __builtin_amdgcn_wmma_f32_16x16x32_bf16(
            false, aq[s].v, false, bk.v, (short)0, s_acc[j], false, false);
      }
    }

    // ---- scale + causal mask ----
    const bool diag = (kt == ktmax);
    float pv[4][8];
#pragma unroll
    for (int j = 0; j < 4; j++)
#pragma unroll
      for (int r = 0; r < 8; r++) {
        float v = s_acc[j][r] * 0.125f;          // 1/sqrt(64)
        if (diag) {
          int kvg = kt * 64 + j * 16 + m;
          int qg  = qrow0 + r + hi * 8;
          if (kvg > qg) v = -3.0e38f;
        }
        pv[j][r] = v;
      }

    // ---- online softmax (rows striped over 16-lane halves) ----
#pragma unroll
    for (int r = 0; r < 8; r++) {
      float v = fmaxf(fmaxf(pv[0][r], pv[1][r]), fmaxf(pv[2][r], pv[3][r]));
#pragma unroll
      for (int off = 1; off < 16; off <<= 1)
        v = fmaxf(v, __shfl_xor(v, off, 32));
      float mnew = fmaxf(mrow[r], v);
      float corr = __expf(mrow[r] - mnew);
      mrow[r] = mnew;
      float rsum = 0.f;
#pragma unroll
      for (int j = 0; j < 4; j++) {
        float p = __expf(pv[j][r] - mnew);
        pv[j][r] = p;
        rsum += p;
      }
#pragma unroll
      for (int off = 1; off < 16; off <<= 1)
        rsum += __shfl_xor(rsum, off, 32);
      lrow[r] = lrow[r] * corr + rsum;
#pragma unroll
      for (int j = 0; j < 4; j++) o[j][r] *= corr;
    }

    // ---- stage P -> LDS (same-wave DS ops are in-order: no barrier) ----
#pragma unroll
    for (int j = 0; j < 4; j++)
#pragma unroll
      for (int r = 0; r < 8; r++)
        sPw[(r + hi * 8) * 64 + j * 16 + m] = f2bf(pv[j][r]);

    // ---- O += P * V ----
#pragma unroll
    for (int s = 0; s < 2; s++) {
      AFrag pa;
      const unsigned short* pp = &sPw[m * 64 + s * 32];
#pragma unroll
      for (int p = 0; p < 8; p++)
        pa.u[p] = *(const unsigned int*)(pp + a_kk(p, hi));
#pragma unroll
      for (int j = 0; j < 4; j++) {
        AFrag bv;
        const unsigned short* vp = &sVt[(j * 16 + m) * 64 + s * 32];
#pragma unroll
        for (int p = 0; p < 8; p++)
          bv.u[p] = *(const unsigned int*)(vp + b_kk(p, hi));
        o[j] = __builtin_amdgcn_wmma_f32_16x16x32_bf16(
            false, pa.v, false, bv.v, (short)0, o[j], false, false);
      }
    }
  }

  // ---- normalize and store ctx (bf16, [B,T,D] with head offset) ----
#pragma unroll
  for (int j = 0; j < 4; j++)
#pragma unroll
    for (int r = 0; r < 8; r++) {
      int tg = qrow0 + r + hi * 8;
      ctxb[((size_t)bb * TSEQ + tg) * D_MODEL + h * HDIM + j * 16 + m] =
          f2bf(o[j][r] / lrow[r]);
    }
}

// ---------------------------------------------------------------------------
// Kernel 3: output projection  out = ctx @ Wo + bo  (fp32 out)
// ---------------------------------------------------------------------------
__global__ __launch_bounds__(256) void out_proj(
    const unsigned short* __restrict__ ctxb,
    const unsigned short* __restrict__ Wot,
    const float* __restrict__ bo,
    float* __restrict__ out)
{
  __shared__ __align__(16) unsigned short sA[128 * 32];
  __shared__ __align__(16) unsigned short sBT[64 * 32];

  const int tid  = threadIdx.x;
  const int wave = tid >> 5;
  const int lane = tid & 31;
  const int m    = lane & 15;
  const int hi   = lane >> 4;

  const int row0 = blockIdx.x * 128;
  const int col0 = blockIdx.y * 64;

  const unsigned ldsA = lds_off(&sA[0]);
  const unsigned ldsB = lds_off(&sBT[0]);

  v8f acc[4];
#pragma unroll
  for (int j = 0; j < 4; j++) acc[j] = (v8f){0.f,0.f,0.f,0.f,0.f,0.f,0.f,0.f};

  for (int k0 = 0; k0 < D_MODEL; k0 += 32) {
    __syncthreads();
#if USE_TDM
    if (tid < 32) {
      tdm_load_2d(ctxb + (size_t)row0 * D_MODEL + k0, ldsA, 32, 128, D_MODEL);
      tdm_load_2d(Wot + (size_t)col0 * D_MODEL + k0, ldsB, 32, 64, D_MODEL);
      if (k0 + 32 < D_MODEL)
        __builtin_prefetch(Wot + (size_t)col0 * D_MODEL + k0 + 32, 0, 1);
      tdm_wait();
    }
#else
    copy_tile_bf16(ctxb + (size_t)row0 * D_MODEL + k0, sA, 32, 128, D_MODEL, tid, 256);
    copy_tile_bf16(Wot + (size_t)col0 * D_MODEL + k0, sBT, 32, 64, D_MODEL, tid, 256);
#endif
    __syncthreads();

    AFrag a;
    const unsigned short* ap = &sA[(wave * 16 + m) * 32];
#pragma unroll
    for (int p = 0; p < 8; p++)
      a.u[p] = *(const unsigned int*)(ap + a_kk(p, hi));
#pragma unroll
    for (int j = 0; j < 4; j++) {
      AFrag b;
      const unsigned short* bp = &sBT[(j * 16 + m) * 32];
#pragma unroll
      for (int p = 0; p < 8; p++)
        b.u[p] = *(const unsigned int*)(bp + b_kk(p, hi));
      acc[j] = __builtin_amdgcn_wmma_f32_16x16x32_bf16(
          false, a.v, false, b.v, (short)0, acc[j], false, false);
    }
  }

#pragma unroll
  for (int j = 0; j < 4; j++)
#pragma unroll
    for (int r = 0; r < 8; r++) {
      int grow = row0 + wave * 16 + r + hi * 8;
      int gcol = col0 + j * 16 + m;
      out[(size_t)grow * D_MODEL + gcol] = acc[j][r] + bo[gcol];
    }
}

// ---------------------------------------------------------------------------
extern "C" void kernel_launch(void* const* d_in, const int* in_sizes, int n_in,
                              void* d_out, int out_size, void* d_ws, size_t ws_size,
                              hipStream_t stream) {
  const float* x  = (const float*)d_in[0];
  const float* Wq = (const float*)d_in[1];
  const float* Wk = (const float*)d_in[2];
  const float* Wv = (const float*)d_in[3];
  const float* Wo = (const float*)d_in[4];
  const float* bo = (const float*)d_in[5];
  float* out = (float*)d_out;

  // workspace layout (bf16 elements):
  //   xb[BT*D] | Wqt|Wkt|Wvt|Wot [D*D each, transposed] | qb|kb|vb|cb [BT*D]
  const size_t ne  = (size_t)BT * D_MODEL;   // 8.39M
  const size_t nw  = (size_t)D_MODEL * D_MODEL;
  unsigned short* xb  = (unsigned short*)d_ws;
  unsigned short* Wqt = xb + ne;
  unsigned short* Wkt = Wqt + nw;
  unsigned short* Wvt = Wkt + nw;
  unsigned short* Wot = Wvt + nw;
  unsigned short* qb  = Wot + nw;
  unsigned short* kb  = qb + ne;
  unsigned short* vb  = kb + ne;
  unsigned short* cb  = vb + ne;

  // 0) one-time precision conversion + weight transposes
  {
    int n4 = (int)(ne / 4);
    cvt_bf16<<<dim3((n4 + 255) / 256), 256, 0, stream>>>(x, xb, n4);
    dim3 tg(D_MODEL / 32, D_MODEL / 32);
    cvt_bf16_T<<<tg, 256, 0, stream>>>(Wq, Wqt);
    cvt_bf16_T<<<tg, 256, 0, stream>>>(Wk, Wkt);
    cvt_bf16_T<<<tg, 256, 0, stream>>>(Wv, Wvt);
    cvt_bf16_T<<<tg, 256, 0, stream>>>(Wo, Wot);
  }
  // 1) Q/K/V projections
  qkv_gemm<<<dim3(BT / 128, (3 * D_MODEL) / 64), 256, 0, stream>>>(
      xb, Wqt, Wkt, Wvt, qb, kb, vb);
  // 2) fused causal attention
  flash_attn<<<dim3(TSEQ / 64, NHEADS, NB), 128, 0, stream>>>(qb, kb, vb, cb);
  // 3) output projection + bias
  out_proj<<<dim3(BT / 128, D_MODEL / 64), 256, 0, stream>>>(cb, Wot, bo, out);
}
